// RelationsNetwork_46943992546197
// MI455X (gfx1250) — matrix operand order repaired and checked by reading, not compiled, not run on hardware
//
#include <hip/hip_runtime.h>
#include <hip/hip_bf16.h>

// ---------------------------------------------------------------------------
// Relations Network, fused WMMA-bf16 pipeline for gfx1250 (MI455X).
//
// Roofline: ~52 GFLOP after factoring g_w1, vs ~3 MB of real data -> compute
// bound; WMMA bf16 (v_wmma_f32_16x16x32_bf16) with fp32 accumulate is the
// throughput path. Pairs tensor is never materialized; the product tile lives
// in LDS (36 KB, reused as the inter-layer h tile).
// ---------------------------------------------------------------------------

typedef __attribute__((ext_vector_type(16))) __bf16 v16bf;
typedef __attribute__((ext_vector_type(8)))  float  v8f;

constexpr int NB   = 32;    // batch
constexpr int SEQ  = 64;    // sequence length (= N_POS)
constexpr int DM   = 256;   // d_model
constexpr int HID  = 256;   // hidden
constexpr int OUTD = 10;
constexpr int DIN  = 258;   // d_model + 2 coords
constexpr int KP1  = 288;   // DIN padded to a multiple of 32 (9 K-steps)

// ---- bf16 helpers (bit-level, RNE) ----------------------------------------
__device__ __forceinline__ float bflo(unsigned u) { return __uint_as_float(u << 16); }
__device__ __forceinline__ float bfhi(unsigned u) { return __uint_as_float(u & 0xFFFF0000u); }
__device__ __forceinline__ unsigned short f2bf(float f) {
  unsigned u = __float_as_uint(f);
  unsigned r = u + 0x7FFFu + ((u >> 16) & 1u);
  return (unsigned short)(r >> 16);
}

// ---- WMMA fragment loaders (ISA 7.12.2 layouts) ---------------------------
struct Pack32 { uint4 lo, hi; };

// A (16x32 bf16): lanes 0-15 -> M=lane, K = k0+{0..7,16..23}; lanes 16-31 add +8.
// Caller passes p = row_ptr + k0 + (hi16 ? 8 : 0).
__device__ __forceinline__ v16bf load_frag_a(const unsigned short* p) {
  Pack32 q;
  q.lo = *(const uint4*)(p);
  q.hi = *(const uint4*)(p + 16);
  return __builtin_bit_cast(v16bf, q);
}
// B (32x16 bf16), weights stored N-major: lanes 0-15 -> K = k0+{0..7,16..23},
// lanes 16-31 add +8. p = rowN_ptr + k0 + (hi16 ? 8 : 0).
__device__ __forceinline__ v16bf load_frag_b(const unsigned short* p) {
  Pack32 q;
  q.lo = *(const uint4*)(p);
  q.hi = *(const uint4*)(p + 16);
  return __builtin_bit_cast(v16bf, q);
}

// ---- prep kernels ---------------------------------------------------------
__global__ void build_xbf(const float* __restrict__ sent,
                          const float* __restrict__ coord,
                          unsigned short* __restrict__ xbf) {
  int row = blockIdx.x;            // b*SEQ + s
  int b = row >> 6, s = row & 63;
  int k = threadIdx.x;             // 0..287
  float v = 0.f;
  if (k < DM)        v = sent[((size_t)b * SEQ + s) * DM + k];
  else if (k < DIN)  v = coord[((size_t)b * SEQ + s) * 2 + (k - DM)];
  xbf[(size_t)row * KP1 + k] = f2bf(v);
}

// dst[n][k] = src[rowOff+k][n] as bf16, zero-padded to KP.
__global__ void transpose_w(const float* __restrict__ w,
                            unsigned short* __restrict__ wt,
                            int K, int N, int KP, int rowOff) {
  int n = blockIdx.x;
  int k = threadIdx.x;             // blockDim == KP
  float v = (k < K) ? w[(size_t)(rowOff + k) * N + n] : 0.f;
  wt[(size_t)n * KP + k] = f2bf(v);
}

// Ajb[b,s,n] = g_b1[n] + x[b,s] @ W1a ; Bi[b,s,n] = x[b,s] @ W1b (fp32).
__global__ void precompute_ab(const float* __restrict__ sent,
                              const float* __restrict__ coord,
                              const float* __restrict__ g_w1,
                              const float* __restrict__ g_b1,
                              float* __restrict__ Ajb,
                              float* __restrict__ Bi,
                              float* __restrict__ rel) {
  __shared__ float xr[DIN];
  int row = blockIdx.x;
  int b = row >> 6, s = row & 63;
  int n = threadIdx.x;             // 256
  for (int k = n; k < DIN; k += 256)
    xr[k] = (k < DM) ? sent[((size_t)b * SEQ + s) * DM + k]
                     : coord[((size_t)b * SEQ + s) * 2 + (k - DM)];
  __syncthreads();
  float a = g_b1[n], c = 0.f;
  for (int k = 0; k < DIN; ++k) {
    float xv = xr[k];
    a += xv * g_w1[(size_t)k * HID + n];
    c += xv * g_w1[(size_t)(DIN + k) * HID + n];
  }
  Ajb[(size_t)row * HID + n] = a;
  Bi[(size_t)row * HID + n]  = c;
  if (s == 0) rel[(size_t)b * HID + n] = 0.f;   // zero accumulator each call
}

// ---- fused 3-layer WMMA chain helpers -------------------------------------
template <int KP>
__device__ __forceinline__ void gemm_tiles(const unsigned short* __restrict__ Abase, // [64][KP] (LDS)
                                           const unsigned short* __restrict__ WT,    // [256][KP] (global)
                                           v8f acc[4][2], int l15, int hi16, int n0) {
  const int koffA = hi16 * 8;
  const int koffB = hi16 * 8;
  for (int kt = 0; kt < KP / 32; ++kt) {
    const int k0 = kt * 32;
    v16bf Af[4];
#pragma unroll
    for (int mt = 0; mt < 4; ++mt)
      Af[mt] = load_frag_a(Abase + (mt * 16 + l15) * KP + k0 + koffA);
    v16bf B0 = load_frag_b(WT + (size_t)(n0 + l15) * KP + k0 + koffB);
    v16bf B1 = load_frag_b(WT + (size_t)(n0 + 16 + l15) * KP + k0 + koffB);
#pragma unroll
    for (int mt = 0; mt < 4; ++mt) {
      acc[mt][0] = __builtin_amdgcn_wmma_f32_16x16x32_bf16(
          false, Af[mt], false, B0, (short)0, acc[mt][0], false, false);
      acc[mt][1] = __builtin_amdgcn_wmma_f32_16x16x32_bf16(
          false, Af[mt], false, B1, (short)0, acc[mt][1], false, false);
    }
  }
}

__device__ __forceinline__ void init_bias(v8f acc[4][2], const float* __restrict__ bias,
                                          int l15, int n0) {
#pragma unroll
  for (int nt = 0; nt < 2; ++nt) {
    float bz = bias[n0 + nt * 16 + l15];
#pragma unroll
    for (int mt = 0; mt < 4; ++mt)
#pragma unroll
      for (int r = 0; r < 8; ++r) acc[mt][nt][r] = bz;
  }
}

// C/D layout: VGPR r -> M = mt*16 + r + hi16*8, N = n0 + nt*16 + (lane&15).
__device__ __forceinline__ void store_relu_bf16(unsigned short* __restrict__ h,
                                                v8f acc[4][2], int l15, int hi16, int n0) {
#pragma unroll
  for (int nt = 0; nt < 2; ++nt) {
    int col = n0 + nt * 16 + l15;
#pragma unroll
    for (int mt = 0; mt < 4; ++mt)
#pragma unroll
      for (int r = 0; r < 8; ++r) {
        int row = mt * 16 + hi16 * 8 + r;
        h[row * HID + col] = f2bf(fmaxf(acc[mt][nt][r], 0.f));
      }
  }
}

// ---- main kernel: one (b, i) strip per workgroup --------------------------
__global__ __launch_bounds__(256) void rn_main(const unsigned short* __restrict__ xbf,
                                               const unsigned short* __restrict__ W1cT,
                                               const unsigned short* __restrict__ W2T,
                                               const unsigned short* __restrict__ W3T,
                                               const float* __restrict__ Ajb,
                                               const float* __restrict__ Bi,
                                               const float* __restrict__ g_b2,
                                               const float* __restrict__ g_b3,
                                               float* __restrict__ rel) {
  extern __shared__ unsigned short sbuf[];   // 64*288 bf16 = 36,864 B; reused as h tile
  int b = blockIdx.x >> 6, i = blockIdx.x & 63;
  int tid = threadIdx.x;

  // Phase 0: product tile p[j][k] = x[b,j][k] * x[b,i][k] (bf16, 2-at-a-time)
  const unsigned* xrow = (const unsigned*)(xbf + (size_t)b * SEQ * KP1);
  const unsigned* xi   = xrow + (size_t)i * (KP1 / 2);
  unsigned* p32 = (unsigned*)sbuf;
  for (int idx = tid; idx < SEQ * (KP1 / 2); idx += 256) {
    int j = idx / (KP1 / 2), kk = idx - j * (KP1 / 2);
    unsigned aj = xrow[(size_t)j * (KP1 / 2) + kk];
    unsigned ai = xi[kk];
    float p0 = bflo(aj) * bflo(ai);
    float p1 = bfhi(aj) * bfhi(ai);
    p32[idx] = (unsigned)f2bf(p0) | ((unsigned)f2bf(p1) << 16);
  }
  __syncthreads();

  int lane = tid & 31, wave = tid >> 5;
  int l15 = lane & 15, hi16 = (lane >> 4) & 1;
  int n0 = wave * 32;                       // this wave's 32 output columns

  const float* AjbB = Ajb + (size_t)b * SEQ * HID;
  const float* BiI  = Bi  + ((size_t)b * SEQ + i) * HID;

  v8f acc[4][2];
  // Layer 1 init: C = Ajb[b,j,:] + Bi[b,i,:]  (bias folded into Ajb)
#pragma unroll
  for (int nt = 0; nt < 2; ++nt) {
    int col = n0 + nt * 16 + l15;
    float bb = BiI[col];
#pragma unroll
    for (int mt = 0; mt < 4; ++mt)
#pragma unroll
      for (int r = 0; r < 8; ++r) {
        int row = mt * 16 + hi16 * 8 + r;
        acc[mt][nt][r] = AjbB[(size_t)row * HID + col] + bb;
      }
  }
  gemm_tiles<KP1>(sbuf, W1cT, acc, l15, hi16, n0);   // + product @ W1c
  __syncthreads();                                    // all sbuf reads done
  store_relu_bf16(sbuf, acc, l15, hi16, n0);          // h1 -> sbuf [64][256]
  __syncthreads();

  init_bias(acc, g_b2, l15, n0);
  gemm_tiles<HID>(sbuf, W2T, acc, l15, hi16, n0);
  __syncthreads();
  store_relu_bf16(sbuf, acc, l15, hi16, n0);          // h2 -> sbuf
  __syncthreads();

  init_bias(acc, g_b3, l15, n0);
  gemm_tiles<HID>(sbuf, W3T, acc, l15, hi16, n0);

  // relu(h3) column sums over this strip's 64 rows -> atomic into rel[b,:]
  float* relB = rel + (size_t)b * HID;
#pragma unroll
  for (int nt = 0; nt < 2; ++nt) {
    float cs = 0.f;
#pragma unroll
    for (int mt = 0; mt < 4; ++mt)
#pragma unroll
      for (int r = 0; r < 8; ++r) cs += fmaxf(acc[mt][nt][r], 0.f);
    cs += __shfl_xor(cs, 16, 32);   // combine M halves across lane groups
    if (lane < 16) atomicAdd(&relB[n0 + nt * 16 + lane], cs);
  }
}

// ---- final f-MLP head (tiny: 32x256 rows) ---------------------------------
__global__ __launch_bounds__(256) void fmlp(const float* __restrict__ rel,
                                            const float* __restrict__ w1, const float* __restrict__ b1,
                                            const float* __restrict__ w2, const float* __restrict__ b2,
                                            const float* __restrict__ w3, const float* __restrict__ b3,
                                            float* __restrict__ out) {
  extern __shared__ float fbuf[];   // 32*256 floats
  int tid = threadIdx.x;
  for (int idx = tid; idx < NB * HID; idx += 256) fbuf[idx] = rel[idx];
  __syncthreads();
  float y[32];
#pragma unroll 1
  for (int t = 0; t < 32; ++t) {
    int o = tid + t * 256, bb = o >> 8, n = o & 255;
    float s = b1[n];
    for (int k = 0; k < HID; ++k) s += fbuf[bb * HID + k] * w1[(size_t)k * HID + n];
    y[t] = fmaxf(s, 0.f);
  }
  __syncthreads();
  for (int t = 0; t < 32; ++t) fbuf[tid + t * 256] = y[t];
  __syncthreads();
#pragma unroll 1
  for (int t = 0; t < 32; ++t) {
    int o = tid + t * 256, bb = o >> 8, n = o & 255;
    float s = b2[n];
    for (int k = 0; k < HID; ++k) s += fbuf[bb * HID + k] * w2[(size_t)k * HID + n];
    y[t] = fmaxf(s, 0.f);
  }
  __syncthreads();
  for (int t = 0; t < 32; ++t) fbuf[tid + t * 256] = y[t];
  __syncthreads();
  for (int o = tid; o < NB * OUTD; o += 256) {
    int bb = o / OUTD, oc = o % OUTD;
    float s = b3[oc];
    for (int k = 0; k < HID; ++k) s += fbuf[bb * HID + k] * w3[(size_t)k * OUTD + oc];
    out[o] = s;
  }
}

// ---- host glue ------------------------------------------------------------
extern "C" void kernel_launch(void* const* d_in, const int* in_sizes, int n_in,
                              void* d_out, int out_size, void* d_ws, size_t ws_size,
                              hipStream_t stream) {
  (void)in_sizes; (void)n_in; (void)out_size; (void)ws_size;
  const float* sent  = (const float*)d_in[0];
  const float* coord = (const float*)d_in[1];
  const float* g_w1  = (const float*)d_in[2];
  const float* g_b1  = (const float*)d_in[3];
  const float* g_w2  = (const float*)d_in[4];
  const float* g_b2  = (const float*)d_in[5];
  const float* g_w3  = (const float*)d_in[6];
  const float* g_b3  = (const float*)d_in[7];
  const float* f_w1  = (const float*)d_in[8];
  const float* f_b1  = (const float*)d_in[9];
  const float* f_w2  = (const float*)d_in[10];
  const float* f_b2  = (const float*)d_in[11];
  const float* f_w3  = (const float*)d_in[12];
  const float* f_b3  = (const float*)d_in[13];

  char* ws = (char*)d_ws;
  // workspace layout (all offsets 256B aligned), total ~5.6 MB
  unsigned short* xbf  = (unsigned short*)(ws);                        // 32*64*288*2 = 1,179,648
  unsigned short* W1cT = (unsigned short*)(ws + 1179648);              // 256*288*2   =   147,456
  unsigned short* W2T  = (unsigned short*)(ws + 1179648 + 147456);     // 256*256*2   =   131,072
  unsigned short* W3T  = (unsigned short*)(ws + 1179648 + 147456 + 131072);
  float* Ajb = (float*)(ws + 1589248);                                 // 32*64*256*4 = 2,097,152
  float* Bi  = (float*)(ws + 1589248 + 2097152);
  float* rel = (float*)(ws + 1589248 + 2 * 2097152);                   // 32*256*4 = 32,768

  build_xbf<<<NB * SEQ, KP1, 0, stream>>>(sent, coord, xbf);
  transpose_w<<<HID, KP1, 0, stream>>>(g_w1, W1cT, DIN, HID, KP1, 2 * DIN);
  transpose_w<<<HID, HID, 0, stream>>>(g_w2, W2T, HID, HID, HID, 0);
  transpose_w<<<HID, HID, 0, stream>>>(g_w3, W3T, HID, HID, HID, 0);
  precompute_ab<<<NB * SEQ, HID, 0, stream>>>(sent, coord, g_w1, g_b1, Ajb, Bi, rel);

  size_t lds_main = (size_t)SEQ * KP1 * sizeof(unsigned short);  // 36,864 B
  rn_main<<<NB * SEQ, 256, lds_main, stream>>>(xbf, W1cT, W2T, W3T, Ajb, Bi, g_b2, g_b3, rel);

  size_t lds_f = (size_t)NB * HID * sizeof(float);               // 32,768 B
  fmlp<<<1, 256, lds_f, stream>>>(rel, f_w1, f_b1, f_w2, f_b2, f_w3, f_b3, (float*)d_out);
}
